// TemporalAttention_12292196401326
// MI455X (gfx1250) — matrix operand order repaired and checked by reading, not compiled, not run on hardware
//
#include <hip/hip_runtime.h>
#include <cstddef>

// ---------------------------------------------------------------------------
// TemporalAttention for MI455X (gfx1250, wave32, WMMA bf16 16x16x32)
//   out = [ V(B,L,H,E) | series(B,H,L,L) | prior(B,H,L,L) | sig4(B,H,L,L) ]
// Store-bandwidth bound (~411 MB of output): all output stores are
// non-temporal so the 40 MB input working set stays resident in the 192 MB L2.
// ---------------------------------------------------------------------------

namespace {

constexpr int Bk = 4;
constexpr int Lk = 1024;
constexpr int Hk = 8;
constexpr int Ek = 64;
constexpr int RSTR = Hk * Ek;      // 512 floats between consecutive l rows
constexpr float kScale = 0.125f;   // 1/sqrt(64)

typedef __bf16 bf16_t;
typedef bf16_t v16bf __attribute__((ext_vector_type(16)));
typedef bf16_t v8bf  __attribute__((ext_vector_type(8)));
typedef float  v8f   __attribute__((ext_vector_type(8)));
typedef float  v4f   __attribute__((ext_vector_type(4)));

// 16 contiguous f32 -> bf16 vector (B-fragment row chunk)
__device__ __forceinline__ v16bf load16bf(const float* p) {
  v16bf r;
#pragma unroll
  for (int j = 0; j < 16; j += 4) {
    float4 x = *(const float4*)(p + j);
    r[j + 0] = (bf16_t)x.x; r[j + 1] = (bf16_t)x.y;
    r[j + 2] = (bf16_t)x.z; r[j + 3] = (bf16_t)x.w;
  }
  return r;
}

// A-matrix (16x32 bf16) fragment from a global f32 row (contiguous k order):
// lane<16  : a[0..15]=row[k0..k0+15]
// lane>=16 : a[0..15]=row[k0+16..k0+31]
__device__ __forceinline__ v16bf loadA_global(const float* row, int k0, int half) {
  const float* p0 = row + k0 + half * 16;
  const float* p1 = p0 + 8;
  float4 x0 = *(const float4*)p0, x1 = *(const float4*)(p0 + 4);
  float4 y0 = *(const float4*)p1, y1 = *(const float4*)(p1 + 4);
  v16bf a;
  a[0]=(bf16_t)x0.x;  a[1]=(bf16_t)x0.y;  a[2]=(bf16_t)x0.z;  a[3]=(bf16_t)x0.w;
  a[4]=(bf16_t)x1.x;  a[5]=(bf16_t)x1.y;  a[6]=(bf16_t)x1.z;  a[7]=(bf16_t)x1.w;
  a[8]=(bf16_t)y0.x;  a[9]=(bf16_t)y0.y;  a[10]=(bf16_t)y0.z; a[11]=(bf16_t)y0.w;
  a[12]=(bf16_t)y1.x; a[13]=(bf16_t)y1.y; a[14]=(bf16_t)y1.z; a[15]=(bf16_t)y1.w;
  return a;
}

// reductions within a 16-lane half-wave (matches C-matrix row striping)
__device__ __forceinline__ float hmax16(float v) {
#pragma unroll
  for (int off = 8; off >= 1; off >>= 1)
    v = fmaxf(v, __shfl_xor(v, off, 32));
  return v;
}
__device__ __forceinline__ float hsum16(float v) {
#pragma unroll
  for (int off = 8; off >= 1; off >>= 1)
    v += __shfl_xor(v, off, 32);
  return v;
}

// S(16 x 64) = Qrows(16xE) * K^T(Ex64) for s-columns [s0, s0+64)
__device__ __forceinline__ void qk_tile(v16bf aq0, v16bf aq1,
                                        const float* kbh, int s0,
                                        int lane16, int half, float t[4][8]) {
#pragma unroll
  for (int cg = 0; cg < 4; ++cg) {
    // B-matrix 32x16: lane n holds K[s0+cg*16+n][e-slice], contiguous in e
    const float* krow = kbh + (size_t)(s0 + cg * 16 + lane16) * RSTR;
    v16bf b0 = load16bf(krow + half * 16);        // e = [0,32)  slice
    v16bf b1 = load16bf(krow + 32 + half * 16);   // e = [32,64) slice
    v8f c = {};
    c = __builtin_amdgcn_wmma_f32_16x16x32_bf16(false, aq0, false, b0,
                                                (short)0, c, false, false);
    c = __builtin_amdgcn_wmma_f32_16x16x32_bf16(false, aq1, false, b1,
                                                (short)0, c, false, false);
#pragma unroll
    for (int v = 0; v < 8; ++v) t[cg][v] = c[v];
  }
}

__device__ __forceinline__ void mask_scale(float t[4][8], int row0, int s0,
                                           int lane16, int half) {
#pragma unroll
  for (int cg = 0; cg < 4; ++cg) {
    const int j = s0 + cg * 16 + lane16;
#pragma unroll
    for (int v = 0; v < 8; ++v) {
      const int i = row0 + v + 8 * half;
      t[cg][v] = (j <= i) ? t[cg][v] * kScale : -__builtin_inff();
    }
  }
}

// online softmax statistics (row max m[], row sum ssum[]) over causal tiles
__device__ __forceinline__ void softmax_stats(v16bf aq0, v16bf aq1,
                                              const float* kbh, int row0,
                                              int ntiles, int lane16, int half,
                                              float m[8], float ssum[8]) {
#pragma unroll
  for (int v = 0; v < 8; ++v) { m[v] = -1.0e30f; ssum[v] = 0.0f; }
  for (int st = 0; st < ntiles; ++st) {
    const int s0 = st * 64;
    float t[4][8];
    qk_tile(aq0, aq1, kbh, s0, lane16, half, t);
    mask_scale(t, row0, s0, lane16, half);
#pragma unroll
    for (int v = 0; v < 8; ++v) {
      float tm = fmaxf(fmaxf(t[0][v], t[1][v]), fmaxf(t[2][v], t[3][v]));
      tm = hmax16(tm);
      const float mn = fmaxf(m[v], tm);
      const float corr = __expf(m[v] - mn);
      float le = __expf(t[0][v] - mn) + __expf(t[1][v] - mn) +
                 __expf(t[2][v] - mn) + __expf(t[3][v] - mn);
      le = hsum16(le);
      ssum[v] = ssum[v] * corr + le;
      m[v] = mn;
    }
  }
}

}  // namespace

// ------------------------- kernel 1: series matrix -------------------------
__global__ __launch_bounds__(256) void series_softmax_kernel(
    const float* __restrict__ Q, const float* __restrict__ K,
    float* __restrict__ series) {
  const int tid = threadIdx.x;
  const int wave = tid >> 5;
  const int ln = tid & 31;
  const int half = ln >> 4;
  const int lane16 = ln & 15;
  const int bh = blockIdx.y;
  const int b = bh >> 3, h = bh & 7;
  const int row0 = blockIdx.x * 128 + wave * 16;

  const float* qbh = Q + (size_t)b * Lk * RSTR + h * Ek;
  const float* kbh = K + (size_t)b * Lk * RSTR + h * Ek;
  float* sbh = series + (size_t)bh * Lk * Lk;

  const float* qrow = qbh + (size_t)(row0 + lane16) * RSTR;
  const v16bf aq0 = loadA_global(qrow, 0, half);
  const v16bf aq1 = loadA_global(qrow, 32, half);

  const int ntiles = ((row0 + 15) >> 6) + 1;  // causal: only tiles reaching diag

  float m[8], ssum[8];
  softmax_stats(aq0, aq1, kbh, row0, ntiles, lane16, half, m, ssum);
  float inv_s[8];
#pragma unroll
  for (int v = 0; v < 8; ++v) inv_s[v] = 1.0f / ssum[v];

  for (int st = 0; st < (Lk / 64); ++st) {
    const int s0 = st * 64;
    if (st < ntiles) {
      float t[4][8];
      qk_tile(aq0, aq1, kbh, s0, lane16, half, t);
      mask_scale(t, row0, s0, lane16, half);
#pragma unroll
      for (int cg = 0; cg < 4; ++cg) {
        const int j = s0 + cg * 16 + lane16;
#pragma unroll
        for (int v = 0; v < 8; ++v) {
          const int i = row0 + v + 8 * half;
          __builtin_nontemporal_store(__expf(t[cg][v] - m[v]) * inv_s[v],
                                      &sbh[(size_t)i * Lk + j]);
        }
      }
    } else {  // fully above the diagonal: softmax is exactly zero
#pragma unroll
      for (int cg = 0; cg < 4; ++cg) {
        const int j = s0 + cg * 16 + lane16;
#pragma unroll
        for (int v = 0; v < 8; ++v) {
          const int i = row0 + v + 8 * half;
          __builtin_nontemporal_store(0.0f, &sbh[(size_t)i * Lk + j]);
        }
      }
    }
  }
}

// --------------------- kernel 2: attention output (P·V) ---------------------
__global__ __launch_bounds__(256) void attn_pv_kernel(
    const float* __restrict__ Q1, const float* __restrict__ K1,
    const float* __restrict__ V, float* __restrict__ outV) {
  __shared__ __align__(16) bf16_t Plds[8 * 16 * 64];  // 2KB per wave, 16KB total

  const int tid = threadIdx.x;
  const int wave = tid >> 5;
  const int ln = tid & 31;
  const int half = ln >> 4;
  const int lane16 = ln & 15;
  const int bh = blockIdx.y;
  const int b = bh >> 3, h = bh & 7;
  const int row0 = blockIdx.x * 128 + wave * 16;

  const float* qbh = Q1 + (size_t)b * Lk * RSTR + h * Ek;
  const float* kbh = K1 + (size_t)b * Lk * RSTR + h * Ek;
  const float* vbh = V  + (size_t)b * Lk * RSTR + h * Ek;
  float* obh = outV + (size_t)b * Lk * RSTR + h * Ek;

  const float* qrow = qbh + (size_t)(row0 + lane16) * RSTR;
  const v16bf aq0 = loadA_global(qrow, 0, half);
  const v16bf aq1 = loadA_global(qrow, 32, half);

  const int ntiles = ((row0 + 15) >> 6) + 1;

  float m[8], ssum[8];
  softmax_stats(aq0, aq1, kbh, row0, ntiles, lane16, half, m, ssum);
  float inv_s[8];
#pragma unroll
  for (int v = 0; v < 8; ++v) inv_s[v] = 1.0f / ssum[v];

  v8f acc[4];
  {
    v8f z = {};
#pragma unroll
    for (int g = 0; g < 4; ++g) acc[g] = z;
  }

  bf16_t* pw = Plds + wave * (16 * 64);  // wave-private P tile (16 rows x 64)

  for (int st = 0; st < ntiles; ++st) {
    const int s0 = st * 64;
    float t[4][8];
    qk_tile(aq0, aq1, kbh, s0, lane16, half, t);
    mask_scale(t, row0, s0, lane16, half);

    // previous iteration's LDS reads must retire before we overwrite the tile
    asm volatile("s_wait_dscnt 0" ::: "memory");
#pragma unroll
    for (int cg = 0; cg < 4; ++cg) {
#pragma unroll
      for (int v = 0; v < 8; ++v) {
        const float p = __expf(t[cg][v] - m[v]) * inv_s[v];  // masked -> 0
        pw[(v + 8 * half) * 64 + cg * 16 + lane16] = (bf16_t)p;
      }
    }
    asm volatile("s_wait_dscnt 0" ::: "memory");

#pragma unroll
    for (int ks = 0; ks < 2; ++ks) {  // contraction over s in steps of 32
      const int k0 = ks * 32;
      const bf16_t* r = pw + lane16 * 64 + k0 + half * 16;
      v8bf lo = *(const v8bf*)(r);        // P[m][k0+16h .. +7]   (16B LDS read)
      v8bf hi = *(const v8bf*)(r + 8);    // P[m][k0+16h+8 .. +15]
      v16bf ap;
#pragma unroll
      for (int j = 0; j < 8; ++j) { ap[j] = lo[j]; ap[8 + j] = hi[j]; }

#pragma unroll
      for (int og = 0; og < 4; ++og) {
        // V B-fragment: lane n holds V[s0+k0(+16)..][og*16+n], stride RSTR
        const float* vp =
            vbh + (size_t)(s0 + k0 + half * 16) * RSTR + og * 16 + lane16;
        v16bf bv;
#pragma unroll
        for (int j = 0; j < 16; ++j) bv[j] = (bf16_t)vp[(size_t)j * RSTR];
        acc[og] = __builtin_amdgcn_wmma_f32_16x16x32_bf16(
            false, ap, false, bv, (short)0, acc[og], false, false);
      }
    }
  }

#pragma unroll
  for (int og = 0; og < 4; ++og) {
    const int d = og * 16 + lane16;
#pragma unroll
    for (int v = 0; v < 8; ++v) {
      const int i = row0 + v + 8 * half;
      __builtin_nontemporal_store(acc[og][v], &obh[(size_t)i * RSTR + d]);
    }
  }
}

// --------------------- kernel 3: Gaussian prior + sig4 ----------------------
__global__ __launch_bounds__(256) void prior_kernel(
    const float* __restrict__ sigma, float* __restrict__ prior,
    float* __restrict__ sig4) {
  const int i = blockIdx.x;
  const int bh = blockIdx.y;
  const int b = bh >> 3, h = bh & 7;
  const float x = sigma[((size_t)b * Lk + i) * Hk + h];
  float sg = 1.0f / (1.0f + __expf(-5.0f * x)) + 1e-5f;
  sg = __expf(sg * 1.0986122886681098f) - 1.0f;  // 3^sg - 1
  const float inv = 0.3989422804014327f / sg;    // 1/sqrt(2*pi)/sg
  const float cc = -0.5f / (sg * sg);
  const size_t base = ((size_t)bh * Lk + i) * Lk;
  const int j0 = threadIdx.x * 4;  // 256 threads x 4 = 1024 columns exactly

  const float d0 = (float)(i - (j0 + 0));
  const float d1 = (float)(i - (j0 + 1));
  const float d2 = (float)(i - (j0 + 2));
  const float d3 = (float)(i - (j0 + 3));
  v4f pr, s4;
  pr[0] = inv * __expf(cc * d0 * d0);
  pr[1] = inv * __expf(cc * d1 * d1);
  pr[2] = inv * __expf(cc * d2 * d2);
  pr[3] = inv * __expf(cc * d3 * d3);
  s4[0] = s4[1] = s4[2] = s4[3] = sg;
  __builtin_nontemporal_store(pr, (v4f*)(prior + base + j0));
  __builtin_nontemporal_store(s4, (v4f*)(sig4 + base + j0));
}

// ------------------------------- launcher -----------------------------------
extern "C" void kernel_launch(void* const* d_in, const int* in_sizes, int n_in,
                              void* d_out, int out_size, void* d_ws,
                              size_t ws_size, hipStream_t stream) {
  (void)in_sizes; (void)n_in; (void)d_ws; (void)ws_size; (void)out_size;
  const float* Q  = (const float*)d_in[0];
  const float* K  = (const float*)d_in[1];
  const float* V  = (const float*)d_in[2];
  const float* Q1 = (const float*)d_in[3];
  const float* K1 = (const float*)d_in[4];
  const float* sg = (const float*)d_in[5];

  float* out    = (float*)d_out;
  float* outV   = out;                                   // [B,L,H,E]
  float* series = out + (size_t)Bk * Lk * Hk * Ek;       // [B,H,L,L]
  float* prior  = series + (size_t)Bk * Hk * Lk * Lk;    // [B,H,L,L]
  float* sig4   = prior + (size_t)Bk * Hk * Lk * Lk;     // [B,H,L,L]

  dim3 gAttn(Lk / 128, Bk * Hk);  // 8 x 32 workgroups, 256 thr (8 waves)
  series_softmax_kernel<<<gAttn, 256, 0, stream>>>(Q, K, series);
  attn_pv_kernel<<<gAttn, 256, 0, stream>>>(Q1, K1, V, outV);
  prior_kernel<<<dim3(Lk, Bk * Hk), 256, 0, stream>>>(sg, prior, sig4);
}
